// DoubleAttentionLayerBiBranch_64321430225069
// MI455X (gfx1250) — matrix operand order repaired and checked by reading, not compiled, not run on hardware
//
#include <hip/hip_runtime.h>
#include <math.h>

// ---------------------------------------------------------------------------
// A2-Net double attention, MI455X (gfx1250, wave32, WMMA bf16 16x16x32).
//
// Key transforms:
//   G = A @ softB^T = wA @ (x @ softB^T) + bA      (softB rows sum to 1)
//   softmax over N is shift invariant -> bB drops out entirely.
// All GEMMs use v_wmma_f32_16x16x32_bf16 with f32 accumulation; operands are
// laid out so each lane issues contiguous 16B/32B global loads matching the
// contiguous k order k = 16*hi + s in both A and B (not the ISA table order). No in-GEMM LDS: working set
// (~120MB ws + 67MB x) largely lives in the 192MB L2 @ effectively >HBM BW.
// ---------------------------------------------------------------------------

typedef __bf16 bf16;
typedef __attribute__((ext_vector_type(16))) __bf16 v16bf;
typedef __attribute__((ext_vector_type(8)))  __bf16 v8bf;
typedef __attribute__((ext_vector_type(8)))  float  v8f;

#define BATCH 8
#define IN_C  512
#define C_M   512
#define C_N   128
#define NPIX  4096   // 64*64

// ---------------------------------------------------------------------------
// Generic WMMA GEMM: D[M x N] = A[M x K] * B[K x N] (+epilogue)
//   A  : bf16 row-major, leading dim lda (K contiguous)
//   BT : bf16, BT[n*ldb + k] == B[k][n]  (K contiguous per column)
// Wave tile: 16(M) x 16*NSUB(N); block = WAVES waves stacked along M.
// EPI: 0 = store f32 ; 1 = store bf16 ; 2 = +bias[m], store bf16
//      3 = +yadd[m][n], store f32
// ---------------------------------------------------------------------------
template<int NSUB, int WAVES, int EPI>
__global__ __launch_bounds__(WAVES * 32)
void wmma_gemm_k(const bf16* __restrict__ A,  long long sA, int lda,
                 const bf16* __restrict__ BT, long long sB, int ldb,
                 void* __restrict__ outp,     long long sO, int ldo,
                 const float* __restrict__ bias,
                 const float* __restrict__ yadd, long long sY,
                 int K)
{
    const int lane = threadIdx.x & 31;
    const int wave = threadIdx.x >> 5;
    const int bId  = blockIdx.z;
    const int hi   = lane >> 4;      // 0: lanes 0-15, 1: lanes 16-31
    const int lw   = lane & 15;

    const int mBase = (blockIdx.y * WAVES + wave) * 16;
    const int nBase = blockIdx.x * (16 * NSUB);

    A  += (long long)bId * sA;
    BT += (long long)bId * sB;

    v8f acc[NSUB];
#pragma unroll
    for (int i = 0; i < NSUB; ++i) acc[i] = (v8f){0.f,0.f,0.f,0.f,0.f,0.f,0.f,0.f};

    // A fragment (16x32 bf16): lane<16 row m=lw holds K k0..k0+15;
    // lane>=16 row m=lw holds K k0+16..k0+31.  -> two 16B loads.
    const bf16* aRow = A + (long long)(mBase + lw) * lda + hi * 16;

    for (int k0 = 0; k0 < K; k0 += 32) {
        union { v16bf v; v8bf h[2]; } af;
        af.h[0] = *(const v8bf*)(aRow + k0);
        af.h[1] = *(const v8bf*)(aRow + k0 + 8);
#pragma unroll
        for (int i = 0; i < NSUB; ++i) {
            // B fragment (32x16 bf16): lane<16 col n=lw holds K k0..k0+15,
            // lane>=16 holds K k0+16..k0+31 -> one contiguous 32B load.
            const bf16* bp = BT + (long long)(nBase + i * 16 + lw) * ldb
                                + k0 + hi * 16;
            v16bf bfr = *(const v16bf*)bp;
            acc[i] = __builtin_amdgcn_wmma_f32_16x16x32_bf16(
                         false, af.v, false, bfr, (short)0, acc[i],
                         false, false);
        }
    }

    // D layout: vgpr r, lane -> D[mBase + r + 8*hi][nBase + i*16 + lw]
    if (EPI == 0 || EPI == 3) {
        float* out = (float*)outp + (long long)bId * sO;
        const float* yb = (EPI == 3) ? (yadd + (long long)bId * sY) : nullptr;
#pragma unroll
        for (int i = 0; i < NSUB; ++i) {
            const int n = nBase + i * 16 + lw;
#pragma unroll
            for (int r = 0; r < 8; ++r) {
                const int m = mBase + hi * 8 + r;
                float v = acc[i][r];
                if (EPI == 3) v += yb[(long long)m * ldo + n];
                out[(long long)m * ldo + n] = v;
            }
        }
    } else {
        bf16* out = (bf16*)outp + (long long)bId * sO;
#pragma unroll
        for (int i = 0; i < NSUB; ++i) {
            const int n = nBase + i * 16 + lw;
#pragma unroll
            for (int r = 0; r < 8; ++r) {
                const int m = mBase + hi * 8 + r;
                float v = acc[i][r];
                if (EPI == 2) v += bias[m];
                out[(long long)m * ldo + n] = (bf16)v;
            }
        }
    }
}

// ---------------------------------------------------------------------------
// Convert weights to bf16: wAbf[512][512]; wBVbf = [wB;wV] stacked [256][512]
// ---------------------------------------------------------------------------
__global__ void cvt_weights_k(const float* __restrict__ wA,
                              const float* __restrict__ wB,
                              const float* __restrict__ wV,
                              bf16* __restrict__ wAbf,
                              bf16* __restrict__ wBVbf)
{
    const int i = blockIdx.x * blockDim.x + threadIdx.x;   // 0 .. 262143
    wAbf[i] = (bf16)wA[i];
    if (i < C_N * IN_C)           wBVbf[i] = (bf16)wB[i];
    else if (i < 2 * C_N * IN_C)  wBVbf[i] = (bf16)wV[i - C_N * IN_C];
}

// ---------------------------------------------------------------------------
// x f32 [b][512][4096] -> Xbf bf16 (same layout) and XbfT bf16 [b][4096][512]
// 32x32 tile transpose through LDS.
// ---------------------------------------------------------------------------
__global__ __launch_bounds__(256)
void cvt_x_k(const float* __restrict__ x,
             bf16* __restrict__ xbf, bf16* __restrict__ xbfT)
{
    __shared__ bf16 t[32][33];
    const int b  = blockIdx.z;
    const int n0 = blockIdx.x * 32;
    const int c0 = blockIdx.y * 32;
    const float* xb = x + (long long)b * IN_C * NPIX;
    bf16* xo  = xbf  + (long long)b * IN_C * NPIX;
    bf16* xto = xbfT + (long long)b * NPIX * IN_C;

#pragma unroll
    for (int rr = 0; rr < 32; rr += 8) {
        const int c = c0 + threadIdx.y + rr;
        const int n = n0 + threadIdx.x;
        const bf16 v = (bf16)xb[(long long)c * NPIX + n];
        xo[(long long)c * NPIX + n] = v;
        t[threadIdx.y + rr][threadIdx.x] = v;
    }
    __syncthreads();
#pragma unroll
    for (int rr = 0; rr < 32; rr += 8) {
        const int n = n0 + threadIdx.y + rr;
        const int c = c0 + threadIdx.x;
        xto[(long long)n * IN_C + c] = t[threadIdx.x][threadIdx.y + rr];
    }
}

// ---------------------------------------------------------------------------
// softB: softmax over N for BV rows 0..127 (bB cancels).  One block per row.
// ---------------------------------------------------------------------------
__global__ __launch_bounds__(256)
void softmax_rows_k(const float* __restrict__ BV, bf16* __restrict__ softB)
{
    const int row = blockIdx.x;             // 0 .. 8*128-1
    const int b = row >> 7, k = row & 127;
    const float* src = BV + ((long long)b * 256 + k) * NPIX;
    bf16* dst = softB + ((long long)b * C_N + k) * NPIX;
    __shared__ float red[256];
    const int t = threadIdx.x;

    float m = -1e30f;
    for (int i = t; i < NPIX; i += 256) m = fmaxf(m, src[i]);
    red[t] = m; __syncthreads();
    for (int s = 128; s; s >>= 1) {
        if (t < s) red[t] = fmaxf(red[t], red[t + s]);
        __syncthreads();
    }
    m = red[0];
    __syncthreads();

    float sum = 0.f;
    for (int i = t; i < NPIX; i += 256) sum += __expf(src[i] - m);
    red[t] = sum; __syncthreads();
    for (int s = 128; s; s >>= 1) {
        if (t < s) red[t] += red[t + s];
        __syncthreads();
    }
    const float inv = 1.f / red[0];
    for (int i = t; i < NPIX; i += 256)
        dst[i] = (bf16)(__expf(src[i] - m) * inv);
}

// ---------------------------------------------------------------------------
// softV: softmax over c_n per pixel (V = BV rows 128..255, +bV), store
// transposed: softVT[b][n][128] bf16.  One thread per pixel; the k-strided
// reads are coalesced across lanes (consecutive n).
// ---------------------------------------------------------------------------
__global__ __launch_bounds__(256)
void softmax_pix_k(const float* __restrict__ BV, const float* __restrict__ bV,
                   bf16* __restrict__ softVT)
{
    const int p = blockIdx.x * blockDim.x + threadIdx.x;  // 0 .. 8*4096-1
    const int b = p >> 12, n = p & 4095;
    const float* V = BV + ((long long)b * 256 + 128) * NPIX + n;

    float m = -1e30f;
    for (int k = 0; k < C_N; ++k) m = fmaxf(m, V[(long long)k * NPIX] + bV[k]);
    float sum = 0.f;
    for (int k = 0; k < C_N; ++k) sum += __expf(V[(long long)k * NPIX] + bV[k] - m);
    const float inv = 1.f / sum;
    bf16* dst = softVT + ((long long)b * NPIX + n) * C_N;
    for (int k = 0; k < C_N; ++k)
        dst[k] = (bf16)(__expf(V[(long long)k * NPIX] + bV[k] - m) * inv);
}

// ---------------------------------------------------------------------------
extern "C" void kernel_launch(void* const* d_in, const int* in_sizes, int n_in,
                              void* d_out, int out_size, void* d_ws, size_t ws_size,
                              hipStream_t stream)
{
    const float* x  = (const float*)d_in[0];
    const float* y  = (const float*)d_in[1];
    const float* wA = (const float*)d_in[2];
    const float* bA = (const float*)d_in[3];
    const float* wB = (const float*)d_in[4];
    // d_in[5] = bB: cancels in softmax over N -> unused.
    const float* wV = (const float*)d_in[6];
    const float* bV = (const float*)d_in[7];
    float* out = (float*)d_out;

    // workspace carve-up (~120.4 MB total)
    char* ws = (char*)d_ws;
    size_t off = 0;
    auto carve = [&](size_t bytes) {
        void* p = ws + off;
        off += (bytes + 255) & ~(size_t)255;
        return p;
    };
    bf16*  Xbf    = (bf16*)carve((size_t)BATCH * IN_C * NPIX * 2);   // 33.6MB
    bf16*  XbfT   = (bf16*)carve((size_t)BATCH * NPIX * IN_C * 2);   // 33.6MB
    float* BV     = (float*)carve((size_t)BATCH * 256  * NPIX * 4);  // 33.6MB
    bf16*  softB  = (bf16*)carve((size_t)BATCH * C_N  * NPIX * 2);   //  8.4MB
    bf16*  softVT = (bf16*)carve((size_t)BATCH * NPIX * C_N  * 2);   //  8.4MB
    bf16*  MT     = (bf16*)carve((size_t)BATCH * C_N  * IN_C * 2);   //  1.0MB
    bf16*  G      = (bf16*)carve((size_t)BATCH * C_M  * C_N  * 2);   //  1.0MB
    bf16*  wAbf   = (bf16*)carve((size_t)C_M * IN_C * 2);
    bf16*  wBVbf  = (bf16*)carve((size_t)2 * C_N * IN_C * 2);

    // 0) weights -> bf16
    cvt_weights_k<<<dim3((C_M * IN_C) / 256), 256, 0, stream>>>(wA, wB, wV, wAbf, wBVbf);

    // 1) x -> bf16 (natural + transposed)
    cvt_x_k<<<dim3(NPIX / 32, IN_C / 32, BATCH), dim3(32, 8), 0, stream>>>(x, Xbf, XbfT);

    // 2) BV[b][256][4096] = [wB;wV] @ x      (M=256, N=4096, K=512)
    wmma_gemm_k<8, 4, 0><<<dim3(NPIX / 128, 256 / 64, BATCH), 128, 0, stream>>>(
        wBVbf, 0, IN_C,
        XbfT, (long long)NPIX * IN_C, IN_C,
        BV, (long long)256 * NPIX, NPIX,
        nullptr, nullptr, 0, IN_C);

    // 3) softB = softmax_N(Bf)
    softmax_rows_k<<<dim3(BATCH * C_N), 256, 0, stream>>>(BV, softB);

    // 4) softVT = softmax_cn(V + bV), transposed
    softmax_pix_k<<<dim3(BATCH * NPIX / 256), 256, 0, stream>>>(BV, bV, softVT);

    // 5) MT[b][128][512] = softB @ x^T       (M=128, N=512, K=4096)
    wmma_gemm_k<4, 4, 1><<<dim3(IN_C / 64, C_N / 64, BATCH), 128, 0, stream>>>(
        softB, (long long)C_N * NPIX, NPIX,
        Xbf, (long long)IN_C * NPIX, NPIX,
        MT, (long long)C_N * IN_C, IN_C,
        nullptr, nullptr, 0, NPIX);

    // 6) G[b][512][128] = wA @ M + bA        (M=512, N=128, K=512)
    wmma_gemm_k<4, 4, 2><<<dim3(C_N / 64, C_M / 64, BATCH), 128, 0, stream>>>(
        wAbf, 0, IN_C,
        MT, (long long)C_N * IN_C, IN_C,
        G, (long long)C_M * C_N, C_N,
        bA, nullptr, 0, IN_C);

    // 7) out = G @ softV + y                 (M=512, N=4096, K=128)
    wmma_gemm_k<8, 4, 3><<<dim3(NPIX / 128, C_M / 64, BATCH), 128, 0, stream>>>(
        G, (long long)C_M * C_N, C_N,
        softVT, (long long)NPIX * C_N, C_N,
        out, (long long)C_M * NPIX, NPIX,
        nullptr, y, (long long)C_M * NPIX, C_N);
}
